// NequIPConvolution_60705067762041
// MI455X (gfx1250) — matrix operand order repaired and checked by reading, not compiled, not run on hardware
//
#include <hip/hip_runtime.h>

// ---------------------------------------------------------------------------
// NequIP convolution for MI455X (gfx1250, wave32, WMMA).
// All dense math -> v_wmma_f32_16x16x32_bf16. Weight matrices are staged in
// LDS *pre-swizzled into per-lane fragment order*, so each 32x16 B-fragment
// load is exactly 2x ds_load_b128 (instead of 16 scalar u16 loads).
// Edge MLP is fused with the tensor product + scatter-add so the [E,256]
// weight tensor is never materialized. W_fc2 is staged through the gfx1250
// async global->LDS DMA (global_load_async_to_lds_b128 + s_wait_asynccnt).
// ---------------------------------------------------------------------------

#define NNODE 20000
#define NEDGE 320000
#define MULQ  64

typedef __attribute__((ext_vector_type(16))) __bf16 v16bf;
typedef __attribute__((ext_vector_type(8)))  float  v8f;
typedef __attribute__((ext_vector_type(4)))  unsigned int u32x4;
typedef __attribute__((ext_vector_type(4)))  int i32x4;

typedef __attribute__((address_space(1))) i32x4 gi32x4;
typedef __attribute__((address_space(3))) i32x4 li32x4;

__device__ __forceinline__ v8f wmma_bf16(v16bf a, v16bf b, v8f c) {
  return __builtin_amdgcn_wmma_f32_16x16x32_bf16(false, a, false, b, (short)0, c,
                                                 false, false);
}

__device__ __forceinline__ float silu_n(float x) {
  return 1.679177f * x / (1.f + __expf(-x));
}

// ---------------------------------------------------------------------------
// Fragment helpers.
// A (16x32 bf16): lanes 0-15 -> M=lane, K 0..7 & 16..23; lanes 16-31 ->
// M=lane-16, K 8..15 & 24..31.
// B (32x16 bf16): lanes 0-15 -> N=lane, K 0..7 & 16..23; lanes 16-31 -> N=lane-16,
// K 8..15 & 24..31.
// C/D (16x16 f32): VGPR r -> M = r + 8*(lane>=16), N = lane&15.
// ---------------------------------------------------------------------------

// Build A from global fp32 (strided), scale folded in.
__device__ __forceinline__ v16bf make_A(const float* base, int row_stride,
                                        int k_stride, float scale) {
  int lane = threadIdx.x & 31;
  int m    = lane & 15;
  int kb   = (lane >> 4) << 3;
  const float* row = base + (size_t)m * row_stride;
  v16bf a;
#pragma unroll
  for (int j = 0; j < 8; ++j) {
    a[j]     = (__bf16)(row[(kb + j)      * k_stride] * scale);
    a[8 + j] = (__bf16)(row[(kb + 16 + j) * k_stride] * scale);
  }
  return a;
}

// A from LDS bf16 row-major [16][lda] (lda multiple of 8): two b128 loads.
__device__ __forceinline__ v16bf load_A_frag(const __bf16* p, int lda) {
  int lane = threadIdx.x & 31;
  int m    = lane & 15;
  int kb   = (lane >> 4) << 3;
  union { u32x4 u[2]; v16bf b; } r;
  r.u[0] = *(const u32x4*)(p + m * lda + kb);
  r.u[1] = *(const u32x4*)(p + m * lda + kb + 16);
  return r.b;
}

// B from a pre-swizzled LDS panel: lane's 16 bf16 are contiguous.
__device__ __forceinline__ v16bf load_B_frag(const __bf16* panel) {
  int lane = threadIdx.x & 31;
  const u32x4* p = (const u32x4*)(panel + lane * 16);
  union { u32x4 u[2]; v16bf b; } r;
  r.u[0] = p[0];
  r.u[1] = p[1];
  return r.b;
}

__device__ __forceinline__ const __bf16* panelp(const __bf16* base, int ntiles,
                                                int kc, int tc) {
  return base + ((size_t)(kc * ntiles + tc) << 9);  // 512 bf16 per panel
}

// Convert a [K][N] row-major fp32 matrix (row stride src_rs, rows >= Kv -> 0)
// into swizzled bf16 B-panels. Panel (kc,tc) holds the 32x16 tile
// (K = kc*32.., N = tc*16..); within a panel, lane l's 16 values are
// contiguous at offset l*16.
__device__ void fill_panels(__bf16* dst, const float* src, int Kv, int K, int N,
                            int src_rs, int tid, int nthreads) {
  int ntiles = N >> 4;
  int total  = K * N;
  for (int i = tid; i < total; i += nthreads) {
    int panel  = i >> 9;
    int within = i & 511;
    int lane   = within >> 4;
    int j      = within & 15;
    int kc     = panel / ntiles;
    int tc     = panel - kc * ntiles;
    int k      = kc * 32 + ((lane >> 4) << 3) + (j & 7) + ((j >> 3) << 4);
    int n      = tc * 16 + (lane & 15);
    dst[i] = (k < Kv) ? (__bf16)src[(size_t)k * src_rs + n] : (__bf16)0.f;
  }
}

__device__ __forceinline__ void store_C(float* base, int row_stride,
                                        int col_stride, v8f c, float scale) {
  int lane = threadIdx.x & 31;
  int n    = lane & 15;
  int mb   = (lane >> 4) << 3;
#pragma unroll
  for (int r = 0; r < 8; ++r)
    base[(size_t)(mb + r) * row_stride + (size_t)n * col_stride] = c[r] * scale;
}

// ---------------------------------------------------------------------------
// Kernel 1: node-side lin1 + self-connection. One wave per 16-node tile.
// ---------------------------------------------------------------------------
__global__ void __launch_bounds__(256) node_pre_kernel(
    const float* __restrict__ node_scalars, const float* __restrict__ node_vectors,
    const float* __restrict__ node_attr,
    const float* __restrict__ Wl1s, const float* __restrict__ Wl1v,
    const float* __restrict__ Wscs, const float* __restrict__ Wscv,
    float* __restrict__ s_lin, float* __restrict__ v_lin,
    float* __restrict__ sc_s, float* __restrict__ sc_v) {
  extern __shared__ char smem_raw[];
  __bf16* wl1s_p = (__bf16*)smem_raw;        // 64x64   -> 8 panels
  __bf16* wl1v_p = wl1s_p + 64 * 64;         // 64x64   -> 8 panels
  __bf16* wscs_p = wl1v_p + 64 * 64;         // 4 x 64x128 -> 4*16 panels
  __bf16* wscv_p = wscs_p + 4 * 64 * 128;    // 4 x 64x64  -> 4*8 panels

  int tid = threadIdx.x;
  fill_panels(wl1s_p, Wl1s, 64, 64, 64, 64, tid, 256);
  fill_panels(wl1v_p, Wl1v, 64, 64, 64, 64, tid, 256);
#pragma unroll
  for (int v = 0; v < 4; ++v) {
    fill_panels(wscs_p + v * 64 * 128, Wscs + v * 128, 64, 64, 128, 4 * 128, tid, 256);
    fill_panels(wscv_p + v * 64 * 64,  Wscv + v * 64,  64, 64, 64,  4 * 64,  tid, 256);
  }
  __syncthreads();

  int tile = blockIdx.x * 8 + (tid >> 5);
  if (tile * 16 >= NNODE) return;  // N/16 == 1250 exact tiles
  int n0   = tile * 16;
  int lane = tid & 31;
  int node = n0 + (lane & 15);

  const float lin_norm = 0.125f;   // 1/sqrt(64)
  const float sc_norm  = 0.0625f;  // 1/sqrt(64*4)

  // ---- lin1 scalars -------------------------------------------------------
  {
    v16bf a0 = make_A(node_scalars + (size_t)n0 * 64,      64, 1, lin_norm);
    v16bf a1 = make_A(node_scalars + (size_t)n0 * 64 + 32, 64, 1, lin_norm);
#pragma unroll
    for (int t = 0; t < 4; ++t) {
      v8f acc = {};
      acc = wmma_bf16(a0, load_B_frag(panelp(wl1s_p, 4, 0, t)), acc);
      acc = wmma_bf16(a1, load_B_frag(panelp(wl1s_p, 4, 1, t)), acc);
      store_C(s_lin + (size_t)n0 * 64 + t * 16, 64, 1, acc, 1.f);
    }
  }
  // ---- lin1 vectors, per component ---------------------------------------
#pragma unroll
  for (int i = 0; i < 3; ++i) {
    v16bf a0 = make_A(node_vectors + (size_t)n0 * 192 + i,          192, 3, lin_norm);
    v16bf a1 = make_A(node_vectors + (size_t)n0 * 192 + 32 * 3 + i, 192, 3, lin_norm);
#pragma unroll
    for (int t = 0; t < 4; ++t) {
      v8f acc = {};
      acc = wmma_bf16(a0, load_B_frag(panelp(wl1v_p, 4, 0, t)), acc);
      acc = wmma_bf16(a1, load_B_frag(panelp(wl1v_p, 4, 1, t)), acc);
      store_C(v_lin + (size_t)n0 * 192 + i + (size_t)t * 16 * 3, 192, 3, acc, 1.f);
    }
  }

  float av[4];
#pragma unroll
  for (int v = 0; v < 4; ++v) av[v] = node_attr[(size_t)node * 4 + v] * sc_norm;

  // ---- self-connection scalars: effective [16,256] @ [256,128] ------------
  {
    v16bf as[4][2];
#pragma unroll
    for (int v = 0; v < 4; ++v) {
      as[v][0] = make_A(node_scalars + (size_t)n0 * 64,      64, 1, av[v]);
      as[v][1] = make_A(node_scalars + (size_t)n0 * 64 + 32, 64, 1, av[v]);
    }
    for (int t = 0; t < 8; ++t) {
      v8f acc = {};
#pragma unroll
      for (int v = 0; v < 4; ++v)
#pragma unroll
        for (int k = 0; k < 2; ++k)
          acc = wmma_bf16(as[v][k],
                          load_B_frag(panelp(wscs_p + v * 64 * 128, 8, k, t)), acc);
      store_C(sc_s + (size_t)n0 * 128 + t * 16, 128, 1, acc, 1.f);
    }
  }
  // ---- self-connection vectors -------------------------------------------
  for (int i = 0; i < 3; ++i) {
    v16bf as[4][2];
#pragma unroll
    for (int v = 0; v < 4; ++v) {
      as[v][0] = make_A(node_vectors + (size_t)n0 * 192 + i,          192, 3, av[v]);
      as[v][1] = make_A(node_vectors + (size_t)n0 * 192 + 32 * 3 + i, 192, 3, av[v]);
    }
#pragma unroll
    for (int t = 0; t < 4; ++t) {
      v8f acc = {};
#pragma unroll
      for (int v = 0; v < 4; ++v)
#pragma unroll
        for (int k = 0; k < 2; ++k)
          acc = wmma_bf16(as[v][k],
                          load_B_frag(panelp(wscv_p + v * 64 * 64, 4, k, t)), acc);
      store_C(sc_v + (size_t)n0 * 192 + i + (size_t)t * 16 * 3, 192, 3, acc, 1.f);
    }
  }
}

// ---------------------------------------------------------------------------
// Kernel 2: fused edge MLP (WMMA) + tensor product + scatter-add.
// One wave per 16-edge tile; 8 waves / block. W_fc2 (64KB fp32) is staged to
// LDS through the gfx1250 async global->LDS DMA when available.
// ---------------------------------------------------------------------------
__global__ void __launch_bounds__(256) edge_kernel(
    const float* __restrict__ edge_sh, const float* __restrict__ edge_emb,
    const int* __restrict__ edge_src, const int* __restrict__ edge_dst,
    const float* __restrict__ Wfc0, const float* __restrict__ Wfc1,
    const float* __restrict__ Wfc2,
    const float* __restrict__ s_lin, const float* __restrict__ v_lin,
    float* __restrict__ s_agg, float* __restrict__ v_agg) {
  extern __shared__ char smem_raw[];
  float*  stage    = (float*)smem_raw;            // 64*256 fp32 (async staging)
  __bf16* wfc0_p   = (__bf16*)(stage + 64 * 256); // 32x64  (K-padded) -> 4 panels
  __bf16* wfc1_p   = wfc0_p + 32 * 64;            // 64x64  -> 8 panels
  __bf16* wfc2_p   = wfc1_p + 64 * 64;            // 64x256 -> 32 panels
  __bf16* hbuf_all = wfc2_p + 64 * 256;           // 8 waves * [16][64]
  int*    esrc_all = (int*)(hbuf_all + 8 * 16 * 64);
  int*    edst_all = esrc_all + 8 * 16;
  float*  y_all    = (float*)(edst_all + 8 * 16);

  int tid = threadIdx.x;

#if __has_builtin(__builtin_amdgcn_global_load_async_to_lds_b128)
  // Async DMA of W_fc2 (fp32) into LDS, 16B per lane per issue; swizzle after.
  for (int i = tid * 4; i < 64 * 256; i += 256 * 4) {
    __builtin_amdgcn_global_load_async_to_lds_b128(
        (gi32x4*)(Wfc2 + i), (li32x4*)(stage + i), 0, 0);
  }
  fill_panels(wfc0_p, Wfc0, 8, 32, 64, 64, tid, 256);
  fill_panels(wfc1_p, Wfc1, 64, 64, 64, 64, tid, 256);
#if __has_builtin(__builtin_amdgcn_s_wait_asynccnt)
  __builtin_amdgcn_s_wait_asynccnt(0);
#else
  asm volatile("s_wait_asynccnt 0x0" ::: "memory");
#endif
  __syncthreads();
  fill_panels(wfc2_p, stage, 64, 64, 256, 256, tid, 256);
#else
  fill_panels(wfc0_p, Wfc0, 8, 32, 64, 64, tid, 256);
  fill_panels(wfc1_p, Wfc1, 64, 64, 64, 64, tid, 256);
  fill_panels(wfc2_p, Wfc2, 64, 64, 256, 256, tid, 256);
#endif
  __syncthreads();

  int wave = tid >> 5, lane = tid & 31;
  int e0 = (blockIdx.x * 8 + wave) * 16;  // E/16 == 20000 exact tiles
  __bf16* hbuf = hbuf_all + (size_t)wave * 16 * 64;
  int*   esrc  = esrc_all + wave * 16;
  int*   edst  = edst_all + wave * 16;
  float* ysh   = y_all + wave * 64;

  if (lane < 16) {
    esrc[lane] = edge_src[e0 + lane];
    edst[lane] = edge_dst[e0 + lane];
#pragma unroll
    for (int j = 0; j < 4; ++j) ysh[lane * 4 + j] = edge_sh[(size_t)(e0 + lane) * 4 + j];
  }
  if (e0 + 128 < NEDGE)
    __builtin_prefetch(edge_emb + (size_t)(e0 + 128) * 8, 0, 1);

  int nloc = lane & 15, mb = (lane >> 4) << 3;

  // ---- layer 0: [16,8(pad32)] @ [32,64] ----------------------------------
  {
    const float n0s = 0.35355339059f;  // 1/sqrt(8)
    int kb = (lane >> 4) << 3;
    int m  = lane & 15;
    v16bf a;
#pragma unroll
    for (int j = 0; j < 8; ++j) {
      int k0 = kb + j;
      a[j]     = (k0 < 8) ? (__bf16)(edge_emb[(size_t)(e0 + m) * 8 + k0] * n0s)
                          : (__bf16)0.f;
      a[8 + j] = (__bf16)0.f;
    }
#pragma unroll
    for (int t = 0; t < 4; ++t) {
      v8f acc = {};
      acc = wmma_bf16(a, load_B_frag(panelp(wfc0_p, 4, 0, t)), acc);
#pragma unroll
      for (int r = 0; r < 8; ++r)  // silu * (1/sqrt(HID)) folded in
        hbuf[(mb + r) * 64 + t * 16 + nloc] = (__bf16)(silu_n(acc[r]) * 0.125f);
    }
  }
  // ---- layer 1: [16,64] @ [64,64] ----------------------------------------
  {
    v16bf a0 = load_A_frag(hbuf, 64);
    v16bf a1 = load_A_frag(hbuf + 32, 64);
#pragma unroll
    for (int t = 0; t < 4; ++t) {
      v8f acc = {};
      acc = wmma_bf16(a0, load_B_frag(panelp(wfc1_p, 4, 0, t)), acc);
      acc = wmma_bf16(a1, load_B_frag(panelp(wfc1_p, 4, 1, t)), acc);
#pragma unroll
      for (int r = 0; r < 8; ++r)
        hbuf[(mb + r) * 64 + t * 16 + nloc] = (__bf16)(silu_n(acc[r]) * 0.125f);
    }
  }
  // ---- layer 2: [16,64] @ [64,256], fused tensor product + scatter -------
  v16bf a0 = load_A_frag(hbuf, 64);
  v16bf a1 = load_A_frag(hbuf + 32, 64);
  const float inv_sqrt3 = 0.57735026919f;
  const float invnn     = 0.0625f;  // 1/N_NEIGH
  for (int t = 0; t < 16; ++t) {
    v8f acc = {};
    acc = wmma_bf16(a0, load_B_frag(panelp(wfc2_p, 16, 0, t)), acc);
    acc = wmma_bf16(a1, load_B_frag(panelp(wfc2_p, 16, 1, t)), acc);
    int g = t >> 2;               // 0:wA 1:wB 2:wC 3:wD (uniform per wave)
    int u = (t & 3) * 16 + nloc;  // channel within group
#pragma unroll
    for (int r = 0; r < 8; ++r) {
      int   le = mb + r;
      int   sn = esrc[le], dn = edst[le];
      float w  = acc[r];
      float y0 = ysh[le * 4 + 0];
      if (g == 0) {
        float sv = s_lin[(size_t)sn * 64 + u];
        unsafeAtomicAdd(&s_agg[(size_t)dn * 128 + u], w * sv * y0 * invnn);
      } else if (g == 1) {
        float sv = s_lin[(size_t)sn * 64 + u];
#pragma unroll
        for (int i = 0; i < 3; ++i)
          unsafeAtomicAdd(&v_agg[(size_t)dn * 384 + u * 3 + i],
                          w * sv * ysh[le * 4 + 1 + i] * invnn);
      } else if (g == 2) {
#pragma unroll
        for (int i = 0; i < 3; ++i) {
          float vv = v_lin[(size_t)sn * 192 + u * 3 + i];
          unsafeAtomicAdd(&v_agg[(size_t)dn * 384 + (64 + u) * 3 + i],
                          w * vv * y0 * invnn);
        }
      } else {
        float dot = 0.f;
#pragma unroll
        for (int i = 0; i < 3; ++i)
          dot += v_lin[(size_t)sn * 192 + u * 3 + i] * ysh[le * 4 + 1 + i];
        unsafeAtomicAdd(&s_agg[(size_t)dn * 128 + 64 + u],
                        w * dot * inv_sqrt3 * invnn);
      }
    }
  }
}

// ---------------------------------------------------------------------------
// Kernel 3: lin2 over aggregated messages, accumulated into sc buffers.
// ---------------------------------------------------------------------------
__global__ void __launch_bounds__(256) node_post_kernel(
    const float* __restrict__ Wl2s, const float* __restrict__ Wl2v,
    const float* __restrict__ s_agg, const float* __restrict__ v_agg,
    float* __restrict__ sc_s, float* __restrict__ sc_v) {
  extern __shared__ char smem_raw[];
  __bf16* w2s_p = (__bf16*)smem_raw;   // 128x128 -> 32 panels
  __bf16* w2v_p = w2s_p + 128 * 128;   // 128x64  -> 16 panels
  int tid = threadIdx.x;
  fill_panels(w2s_p, Wl2s, 128, 128, 128, 128, tid, 256);
  fill_panels(w2v_p, Wl2v, 128, 128, 64, 64, tid, 256);
  __syncthreads();

  int tile = blockIdx.x * 8 + (tid >> 5);
  if (tile * 16 >= NNODE) return;
  int n0   = tile * 16;
  int lane = tid & 31;
  int nloc = lane & 15, mb = (lane >> 4) << 3;
  const float norm = 0.08838834765f;  // 1/sqrt(128)

  // s path: [16,128] @ [128,128]
  {
    v16bf a[4];
#pragma unroll
    for (int k = 0; k < 4; ++k)
      a[k] = make_A(s_agg + (size_t)n0 * 128 + k * 32, 128, 1, 1.f);
    for (int t = 0; t < 8; ++t) {
      v8f acc = {};
#pragma unroll
      for (int k = 0; k < 4; ++k)
        acc = wmma_bf16(a[k], load_B_frag(panelp(w2s_p, 8, k, t)), acc);
#pragma unroll
      for (int r = 0; r < 8; ++r) {
        size_t idx = (size_t)(n0 + mb + r) * 128 + t * 16 + nloc;
        sc_s[idx] += acc[r] * norm;
      }
    }
  }
  // v path: per component, [16,128] @ [128,64]
  for (int i = 0; i < 3; ++i) {
    v16bf a[4];
#pragma unroll
    for (int k = 0; k < 4; ++k)
      a[k] = make_A(v_agg + (size_t)n0 * 384 + k * 32 * 3 + i, 384, 3, 1.f);
#pragma unroll
    for (int t = 0; t < 4; ++t) {
      v8f acc = {};
#pragma unroll
      for (int k = 0; k < 4; ++k)
        acc = wmma_bf16(a[k], load_B_frag(panelp(w2v_p, 4, k, t)), acc);
#pragma unroll
      for (int r = 0; r < 8; ++r) {
        size_t idx = (size_t)(n0 + mb + r) * 192 + (t * 16 + nloc) * 3 + i;
        sc_v[idx] += acc[r] * norm;
      }
    }
  }
}

// ---------------------------------------------------------------------------
// Kernel 4: gated nonlinearity -> output [N, 64 + 64*3]
// ---------------------------------------------------------------------------
__global__ void __launch_bounds__(256) gate_kernel(const float* __restrict__ sc_s,
                                                   const float* __restrict__ sc_v,
                                                   float* __restrict__ out) {
  int idx = blockIdx.x * blockDim.x + threadIdx.x;
  if (idx >= NNODE * MULQ) return;
  int n = idx >> 6, u = idx & 63;
  float gs = sc_s[(size_t)n * 128 + u];
  float gv = sc_s[(size_t)n * 128 + 64 + u];
  float* o = out + (size_t)n * 256;
  o[u] = silu_n(gs);
  float g = silu_n(gv);
#pragma unroll
  for (int i = 0; i < 3; ++i)
    o[64 + u * 3 + i] = g * sc_v[(size_t)n * 192 + u * 3 + i];
}

// ---------------------------------------------------------------------------
extern "C" void kernel_launch(void* const* d_in, const int* in_sizes, int n_in,
                              void* d_out, int out_size, void* d_ws, size_t ws_size,
                              hipStream_t stream) {
  const float* node_scalars = (const float*)d_in[0];
  const float* node_vectors = (const float*)d_in[1];
  const float* node_attr    = (const float*)d_in[2];
  const float* edge_sh      = (const float*)d_in[3];
  const float* edge_emb     = (const float*)d_in[4];
  const int*   edge_src     = (const int*)d_in[5];
  const int*   edge_dst     = (const int*)d_in[6];
  const float* Wl1s = (const float*)d_in[7];
  const float* Wl1v = (const float*)d_in[8];
  const float* Wfc0 = (const float*)d_in[9];
  const float* Wfc1 = (const float*)d_in[10];
  const float* Wfc2 = (const float*)d_in[11];
  const float* Wl2s = (const float*)d_in[12];
  const float* Wl2v = (const float*)d_in[13];
  const float* Wscs = (const float*)d_in[14];
  const float* Wscv = (const float*)d_in[15];

  // workspace carve (fp32): 20000*(64+192+128+192+128+384) = ~87 MB
  float* s_lin = (float*)d_ws;
  float* v_lin = s_lin + (size_t)NNODE * 64;
  float* sc_s  = v_lin + (size_t)NNODE * 192;
  float* sc_v  = sc_s + (size_t)NNODE * 128;
  float* s_agg = sc_v + (size_t)NNODE * 192;
  float* v_agg = s_agg + (size_t)NNODE * 128;

  (void)hipMemsetAsync(s_agg, 0, (size_t)NNODE * 512 * sizeof(float), stream);

  size_t lds_pre = (size_t)(64 * 64 * 2 + 4 * 64 * 128 + 4 * 64 * 64) * sizeof(__bf16);
  node_pre_kernel<<<(NNODE / 16 + 7) / 8, 256, lds_pre, stream>>>(
      node_scalars, node_vectors, node_attr, Wl1s, Wl1v, Wscs, Wscv,
      s_lin, v_lin, sc_s, sc_v);

  size_t lds_edge = (size_t)(64 * 256) * sizeof(float) +
                    (size_t)(32 * 64 + 64 * 64 + 64 * 256 + 8 * 16 * 64) * sizeof(__bf16) +
                    (size_t)(8 * 16 * 2) * sizeof(int) + (size_t)(8 * 64) * sizeof(float);
  edge_kernel<<<NEDGE / 128, 256, lds_edge, stream>>>(
      edge_sh, edge_emb, edge_src, edge_dst, Wfc0, Wfc1, Wfc2,
      s_lin, v_lin, s_agg, v_agg);

  size_t lds_post = (size_t)(128 * 128 + 128 * 64) * sizeof(__bf16);
  node_post_kernel<<<(NNODE / 16 + 7) / 8, 256, lds_post, stream>>>(
      Wl2s, Wl2v, s_agg, v_agg, sc_s, sc_v);

  gate_kernel<<<(NNODE * MULQ + 255) / 256, 256, 0, stream>>>(
      sc_s, sc_v, (float*)d_out);
}
